// Net_14405320311195
// MI455X (gfx1250) — matrix operand order repaired and checked by reading, not compiled, not run on hardware
//
#include <hip/hip_runtime.h>
#include <hip/hip_bf16.h>

typedef __attribute__((ext_vector_type(16))) __bf16 v16bf;
typedef __attribute__((ext_vector_type(8)))  float  v8f;

// Fragment layouts (per 16x16x32 WMMA step, wave32):
//  A (16-bit): lane L (row = L&15, h = L>>4) carries elems j=0..15 with
//              K = ks*32 + 8h + (j&7) + 16*(j>>3)
//  B (16-bit): lane L (col = L&15, h = L>>4) carries elems j=0..15 with
//              K = ks*32 + 8h + (j&7) + 16*(j>>3)
// Packed storage: frag[tile][ks][lane][j] -> 32 contiguous bytes per lane.

// ---------------- degree / norm ----------------

__global__ void k_init_deg(float* deg, int n) {
    int i = blockIdx.x * blockDim.x + threadIdx.x;
    if (i < n) deg[i] = 1.0f;   // self-loop contributes 1
}

__global__ void k_deg_accum(const long long* __restrict__ dst, float* __restrict__ deg, long long E) {
    long long i = (long long)blockIdx.x * blockDim.x + threadIdx.x;
    if (i < E) unsafeAtomicAdd(&deg[dst[i]], 1.0f);
}

__global__ void k_rsqrt(float* d, int n) {
    int i = blockIdx.x * blockDim.x + threadIdx.x;
    if (i < n) d[i] = rsqrtf(d[i]);  // deg >= 1 always
}

// ---------------- packing kernels ----------------

// Pack features f32[Nn][128] -> A-fragment layout bf16 (output-side indexed, coalesced writes)
__global__ void k_pack_a_f32(const float* __restrict__ in, __bf16* __restrict__ Apk, long long total) {
    long long i = (long long)blockIdx.x * blockDim.x + threadIdx.x;
    if (i >= total) return;
    int j    = (int)(i & 15);
    int lane = (int)((i >> 4) & 31);
    int ks   = (int)((i >> 9) & 3);
    long long mtile = i >> 11;
    int h  = lane >> 4;
    int kk = ks * 32 + (h << 3) + (j & 7) + ((j >> 3) << 4);
    long long r = mtile * 16 + (lane & 15);
    Apk[i] = (__bf16)in[r * 128 + kk];
}

// Pack weights f32[128][C] -> B-fragment layout bf16, zero-padded to Cpad columns
__global__ void k_pack_b(const float* __restrict__ W, __bf16* __restrict__ Bpk, int C, int total) {
    int i = blockIdx.x * blockDim.x + threadIdx.x;
    if (i >= total) return;
    int j     = i & 15;
    int lane  = (i >> 4) & 31;
    int ks    = (i >> 9) & 3;
    int ntile = i >> 11;
    int kk = ks * 32 + ((lane >> 4) << 3) + (j & 7) + ((j >> 3) << 4);
    int n  = ntile * 16 + (lane & 15);
    Bpk[i] = (n < C) ? (__bf16)W[kk * C + n] : (__bf16)0.0f;
}

__global__ void k_zero_f32(float* p, long long n) {
    long long i = (long long)blockIdx.x * blockDim.x + threadIdx.x;
    if (i < n) p[i] = 0.0f;
}

// ---------------- WMMA GEMM ----------------
// D[M x Cpad] = A[M x 128] * B[128 x Cpad], both in packed-fragment bf16.
// NT = number of 16-wide N tiles (compile-time). 16 waves/block, one M-tile per wave.

template <int NT>
__global__ void k_gemm_wmma(const __bf16* __restrict__ Apk, const __bf16* __restrict__ Bpk,
                            float* __restrict__ D, int Mtiles) {
    constexpr int Cpad = NT * 16;
    __shared__ __bf16 sB[NT * 2048];   // NT * 4 ks * 32 lanes * 16 elems

    // cooperative stage of packed B into LDS (all threads reach the barrier)
    {
        constexpr int chunks = NT * 256;           // 16-byte chunks
        const uint4* s16 = (const uint4*)Bpk;
        uint4*       d16 = (uint4*)sB;
        for (int t = threadIdx.x; t < chunks; t += 512) d16[t] = s16[t];
    }
    __syncthreads();

    int lane  = threadIdx.x & 31;
    int wave  = threadIdx.x >> 5;
    int mtile = blockIdx.x * 16 + wave;
    if (mtile >= Mtiles) return;

    // all 4 A fragments for this wave's M-tile (2x b128 loads each, contiguous)
    v16bf a[4];
    #pragma unroll
    for (int ks = 0; ks < 4; ++ks)
        a[ks] = *(const v16bf*)(Apk + ((long long)(mtile * 4 + ks) * 32 + lane) * 16);

    int rbase = mtile * 16 + ((lane >> 4) << 3);
    float* drow = D + (long long)rbase * Cpad + (lane & 15);

    #pragma unroll
    for (int nt = 0; nt < NT; ++nt) {
        v8f c = {};
        #pragma unroll
        for (int ks = 0; ks < 4; ++ks) {
            v16bf b = *(const v16bf*)(sB + ((nt * 4 + ks) * 32 + lane) * 16);
            c = __builtin_amdgcn_wmma_f32_16x16x32_bf16(false, a[ks], false, b,
                                                        (short)0, c, false, false);
        }
        #pragma unroll
        for (int v = 0; v < 8; ++v)
            drow[v * Cpad + nt * 16] = c[v];   // immediate offsets off one base
    }
}

// ---------------- scatter-add aggregation (edges + self-loops) ----------------
// One wave per message; lane handles 4 consecutive channels (b128 row read + 4 f32 atomics).

template <int CPAD>
__global__ void k_scatter(const float* __restrict__ h, const long long* __restrict__ src,
                          const long long* __restrict__ dst, const float* __restrict__ dinv,
                          float* __restrict__ agg, long long E, int Nn) {
    long long w = (long long)blockIdx.x * (blockDim.x >> 5) + (threadIdx.x >> 5);
    int lane = threadIdx.x & 31;
    long long total = E + Nn;
    if (w >= total) return;
    long long s, d;
    if (w < E) { s = src[w]; d = dst[w]; } else { s = d = w - E; }
    float nrm = dinv[s] * dinv[d];
    int c0 = lane * 4;
    if (c0 >= CPAD) return;
    const float4 hv = *(const float4*)(h + s * (long long)CPAD + c0);
    float* ar = agg + d * (long long)CPAD + c0;
    unsafeAtomicAdd(ar + 0, hv.x * nrm);
    unsafeAtomicAdd(ar + 1, hv.y * nrm);
    unsafeAtomicAdd(ar + 2, hv.z * nrm);
    unsafeAtomicAdd(ar + 3, hv.w * nrm);
}

// ---------------- epilogues ----------------

// layers 1,2: next A-fragments = pack(bf16(relu(agg + b))), C == Cpad == 128
__global__ void k_bias_relu_pack(const float* __restrict__ agg, const float* __restrict__ b,
                                 __bf16* __restrict__ Apk, long long total) {
    long long i = (long long)blockIdx.x * blockDim.x + threadIdx.x;
    if (i >= total) return;
    int j    = (int)(i & 15);
    int lane = (int)((i >> 4) & 31);
    int ks   = (int)((i >> 9) & 3);
    long long mtile = i >> 11;
    int h  = lane >> 4;
    int kk = ks * 32 + (h << 3) + (j & 7) + ((j >> 3) << 4);
    long long r = mtile * 16 + (lane & 15);
    float v = agg[r * 128 + kk] + b[kk];
    Apk[i] = (__bf16)fmaxf(v, 0.0f);
}

// layer 3: out[i, 0..C) = log_softmax(agg[i, 0..C) + b), one wave per node (C=47, Cpad=48)
__global__ void k_logsoftmax(const float* __restrict__ agg, const float* __restrict__ b,
                             float* __restrict__ out, int Nn, int C, int Cpad) {
    int wave = threadIdx.x >> 5, lane = threadIdx.x & 31;
    int node = blockIdx.x * (blockDim.x >> 5) + wave;
    if (node >= Nn) return;
    const float* r = agg + (long long)node * Cpad;
    float v0 = (lane < C) ? r[lane] + b[lane] : -__builtin_inff();
    int   c1 = lane + 32;
    float v1 = (c1 < C)   ? r[c1]   + b[c1]   : -__builtin_inff();
    float m = fmaxf(v0, v1);
    #pragma unroll
    for (int o = 16; o > 0; o >>= 1) m = fmaxf(m, __shfl_xor(m, o, 32));
    float e = 0.0f;
    if (lane < C) e += expf(v0 - m);
    if (c1 < C)   e += expf(v1 - m);
    #pragma unroll
    for (int o = 16; o > 0; o >>= 1) e += __shfl_xor(e, o, 32);
    float ls = logf(e);
    if (lane < C) out[(long long)node * C + lane] = v0 - m - ls;
    if (c1 < C)   out[(long long)node * C + c1]   = v1 - m - ls;
}

// ---------------- host orchestration ----------------

static inline unsigned cdiv(long long a, long long b) { return (unsigned)((a + b - 1) / b); }

extern "C" void kernel_launch(void* const* d_in, const int* in_sizes, int n_in,
                              void* d_out, int out_size, void* d_ws, size_t ws_size,
                              hipStream_t stream) {
    const float*     x  = (const float*)d_in[0];
    const long long* ei = (const long long*)d_in[1];
    const float* W1 = (const float*)d_in[2]; const float* b1 = (const float*)d_in[3];
    const float* W2 = (const float*)d_in[4]; const float* b2 = (const float*)d_in[5];
    const float* W3 = (const float*)d_in[6]; const float* b3 = (const float*)d_in[7];

    const int Cin = 128;
    const int Nn  = in_sizes[0] / Cin;            // 100000
    const long long E = in_sizes[1] / 2;          // 1.6M
    const int Cout = in_sizes[6] / 128;           // 47
    const int CoutPad = 48;                        // 3 tiles of 16
    float* out = (float*)d_out;

    const long long* src = ei;
    const long long* dst = ei + E;

    char* ws = (char*)d_ws;
    size_t off = 0;
    auto carve = [&](size_t bytes) -> char* {
        char* p = ws + off;
        off = (off + bytes + 255) & ~(size_t)255;
        return p;
    };
    float*  dinv = (float*)carve((size_t)Nn * 4);
    __bf16* Apk  = (__bf16*)carve((size_t)Nn * 128 * 2);
    __bf16* Bpk  = (__bf16*)carve((size_t)128 * 128 * 2);
    float*  h    = (float*)carve((size_t)Nn * 128 * 4);
    float*  agg  = (float*)carve((size_t)Nn * 128 * 4);

    const int T = 256;

    k_init_deg <<<cdiv(Nn, T), T, 0, stream>>>(dinv, Nn);
    k_deg_accum<<<cdiv(E, T),  T, 0, stream>>>(dst, dinv, E);
    k_rsqrt    <<<cdiv(Nn, T), T, 0, stream>>>(dinv, Nn);

    const long long featTotal = (long long)Nn * 128;
    k_pack_a_f32<<<cdiv(featTotal, T), T, 0, stream>>>(x, Apk, featTotal);

    const int Mtiles = (Nn + 15) / 16;            // 6250 (exact: 100000/16)
    const long long totalMsg = E + Nn;

    // ---- layer 1 (128 -> 128) ----
    k_pack_b<<<cdiv(8 * 2048, T), T, 0, stream>>>(W1, Bpk, 128, 8 * 2048);
    k_gemm_wmma<8><<<cdiv(Mtiles, 16), 512, 0, stream>>>(Apk, Bpk, h, Mtiles);
    k_zero_f32<<<cdiv((long long)Nn * 128, T), T, 0, stream>>>(agg, (long long)Nn * 128);
    k_scatter<128><<<cdiv(totalMsg, 8), T, 0, stream>>>(h, src, dst, dinv, agg, E, Nn);
    k_bias_relu_pack<<<cdiv(featTotal, T), T, 0, stream>>>(agg, b1, Apk, featTotal);

    // ---- layer 2 (128 -> 128) ----
    k_pack_b<<<cdiv(8 * 2048, T), T, 0, stream>>>(W2, Bpk, 128, 8 * 2048);
    k_gemm_wmma<8><<<cdiv(Mtiles, 16), 512, 0, stream>>>(Apk, Bpk, h, Mtiles);
    k_zero_f32<<<cdiv((long long)Nn * 128, T), T, 0, stream>>>(agg, (long long)Nn * 128);
    k_scatter<128><<<cdiv(totalMsg, 8), T, 0, stream>>>(h, src, dst, dinv, agg, E, Nn);
    k_bias_relu_pack<<<cdiv(featTotal, T), T, 0, stream>>>(agg, b2, Apk, featTotal);

    // ---- layer 3 (128 -> 47, padded to 48) ----
    k_pack_b<<<cdiv(3 * 2048, T), T, 0, stream>>>(W3, Bpk, Cout, 3 * 2048);
    k_gemm_wmma<3><<<cdiv(Mtiles, 16), 512, 0, stream>>>(Apk, Bpk, h, Mtiles);
    k_zero_f32<<<cdiv((long long)Nn * CoutPad, T), T, 0, stream>>>(agg, (long long)Nn * CoutPad);
    k_scatter<48><<<cdiv(totalMsg, 8), T, 0, stream>>>(h, src, dst, dinv, agg, E, Nn);
    k_logsoftmax<<<cdiv(Nn, 8), T, 0, stream>>>(agg, b3, out, Nn, Cout, CoutPad);
}
